// QuantLowRankAdapter_60258391162946
// MI455X (gfx1250) — matrix operand order repaired and checked by reading, not compiled, not run on hardware
//
#include <hip/hip_runtime.h>
#include <hip/hip_bf16.h>

typedef __attribute__((ext_vector_type(16))) __bf16          v16bf;
typedef __attribute__((ext_vector_type(16))) unsigned short  v16u;
typedef __attribute__((ext_vector_type(8)))  float           v8f;

#define T_TOK 16384
#define D_DIM 4096
#define R_RANK 64

union BF16Frag { v16u u; v16bf b; };

// Native f32 -> bf16 (RNE); lowers to v_cvt_pk_bf16_f32.
__device__ __forceinline__ v16bf pack_bf16(float4 a0, float4 a1, float4 a2, float4 a3) {
  v16bf r;
  r[0]  = (__bf16)a0.x; r[1]  = (__bf16)a0.y; r[2]  = (__bf16)a0.z; r[3]  = (__bf16)a0.w;
  r[4]  = (__bf16)a1.x; r[5]  = (__bf16)a1.y; r[6]  = (__bf16)a1.z; r[7]  = (__bf16)a1.w;
  r[8]  = (__bf16)a2.x; r[9]  = (__bf16)a2.y; r[10] = (__bf16)a2.z; r[11] = (__bf16)a2.w;
  r[12] = (__bf16)a3.x; r[13] = (__bf16)a3.y; r[14] = (__bf16)a3.z; r[15] = (__bf16)a3.w;
  return r;
}

__device__ __forceinline__ float4 add4(float4 a, float4 b) {
  return make_float4(a.x + b.x, a.y + b.y, a.z + b.z, a.w + b.w);
}

// ---------------------------------------------------------------------------
// Dequantize a[64,4096] into bf16 B-fragments for gemm1.
// B-fragment layout (32x16 bf16, wave32): lane l -> column N = l%16,
// K = (l>=16)*8 + e + (e>=8)*8, e in [0,16).  Fragment index = (nt*128 + ks).
// ---------------------------------------------------------------------------
__global__ __launch_bounds__(256) void deq_a_kernel(const int* __restrict__ qv,
                                                    const float* __restrict__ qs,
                                                    unsigned short* __restrict__ frag) {
  int tid  = blockIdx.x * 256 + threadIdx.x;   // 0..16383
  int lane = tid & 31;
  int ks   = (tid >> 5) & 127;
  int nt   = tid >> 12;
  int n_g  = nt * 16 + (lane & 15);
  int k0   = ks * 32 + ((lane >> 4) << 3);
  const int* vp = qv + n_g * D_DIM + k0;
  float s = qs[n_g * (D_DIM / 32) + (k0 >> 5)];
  v16bf f;
#pragma unroll
  for (int e = 0; e < 16; ++e) f[e] = (__bf16)((float)vp[e + (e & 8)] * s);
  BF16Frag w; w.b = f;
  *(v16u*)(frag + (size_t)tid * 16) = w.u;
}

// ---------------------------------------------------------------------------
// Dequantize b[4096,64] into bf16 B-fragments for gemm2.
// Fragment index = (nt*2 + ks), nt in [0,256), ks in [0,2).
// ---------------------------------------------------------------------------
__global__ __launch_bounds__(256) void deq_b_kernel(const int* __restrict__ qv,
                                                    const float* __restrict__ qs,
                                                    unsigned short* __restrict__ frag) {
  int tid  = blockIdx.x * 256 + threadIdx.x;   // 0..16383
  int lane = tid & 31;
  int ks   = (tid >> 5) & 1;
  int nt   = tid >> 6;
  int n_g  = nt * 16 + (lane & 15);
  int k0   = ks * 32 + ((lane >> 4) << 3);
  const int* vp = qv + n_g * R_RANK + k0;
  float s = qs[n_g * (R_RANK / 32) + (k0 >> 5)];
  v16bf f;
#pragma unroll
  for (int e = 0; e < 16; ++e) f[e] = (__bf16)((float)vp[e + (e & 8)] * s);
  BF16Frag w; w.b = f;
  *(v16u*)(frag + (size_t)tid * 16) = w.u;
}

// ---------------------------------------------------------------------------
// GEMM1: partial h slices.  512 blocks x 4 waves: block = (8 M-tiles, one
// 1024-wide K quarter).  B-fragments staged in LDS (64 KB chunks shared by
// all 4 waves; ds_load latency replaces L2 latency at the WMMAs).  Each wave
// owns 2 M-tiles so every fragment feeds 2 WMMAs.  x is ping-pong double-
// buffered straight from HBM (one full 128B line per row per K step).
// ---------------------------------------------------------------------------
__global__ __launch_bounds__(128) void gemm1_kernel(const float* __restrict__ x,
                                                    const unsigned short* __restrict__ afrag,
                                                    float* __restrict__ hpart) {
  __shared__ uint4 lds_b[4096];                 // 64 KB: [nt:4][ks:16][lane:32][32B]
  int bid  = blockIdx.x;                        // 0..511
  int kq   = bid & 3;                           // K quarter
  int mg   = bid >> 2;                          // 0..127 -> 8 M-tiles
  int wid  = threadIdx.x >> 5;                  // 0..3
  int lane = threadIdx.x & 31;
  int row  = lane & 15;
  int khalf = lane >> 4;
  int mt0 = mg * 8 + wid * 2, mt1 = mt0 + 1;
  const float* xr0 = x + (size_t)(mt0 * 16 + row) * D_DIM + kq * 1024 + khalf * 8;
  const float* xr1 = x + (size_t)(mt1 * 16 + row) * D_DIM + kq * 1024 + khalf * 8;
  const v16u* ldsv = (const v16u*)lds_b;        // nt stride = 512 v16u

  v8f a00 = {}, a01 = {}, a02 = {}, a03 = {};   // mt0 accumulators
  v8f a10 = {}, a11 = {}, a12 = {}, a13 = {};   // mt1 accumulators

  for (int chunk = 0; chunk < 2; ++chunk) {
    __syncthreads();  // previous chunk fully consumed before overwrite
#pragma unroll
    for (int nt = 0; nt < 4; ++nt) {
      const uint4* s = (const uint4*)((const v16u*)afrag +
                                      (size_t)(nt * 128 + kq * 32 + chunk * 16) * 32);
      uint4* d = lds_b + nt * 1024;
      for (int i = threadIdx.x; i < 1024; i += 128) d[i] = s[i];
    }
    __syncthreads();

    const float* cx0 = xr0 + chunk * 512;
    const float* cx1 = xr1 + chunk * 512;

    // Prologue: stage local step 0 into buffer 0 (both M tiles).
    const float4* p0 = (const float4*)cx0;
    const float4* p1 = (const float4*)cx1;
    float4 u0a = p0[0], u0b = p0[1], u0c = p0[4], u0d = p0[5];
    float4 v0a = p1[0], v0b = p1[1], v0c = p1[4], v0d = p1[5];

    for (int ksl = 0; ksl < 16; ksl += 2) {
      int k1 = ksl + 1;          // < 16
      int k2 = (ksl + 2) & 15;   // wrap; tail reload harmless, branch-free

      // Prefetch step k1 into buffer 1.
      const float4* q0 = (const float4*)(cx0 + k1 * 32);
      const float4* q1 = (const float4*)(cx1 + k1 * 32);
      float4 u1a = q0[0], u1b = q0[1], u1c = q0[4], u1d = q0[5];
      float4 v1a = q1[0], v1b = q1[1], v1c = q1[4], v1d = q1[5];

      {  // Compute step ksl from buffer 0; B fragments from LDS.
        const v16u* lp = ldsv + ksl * 32 + lane;
        BF16Frag f0, f1, f2, f3;
        f0.u = lp[0]; f1.u = lp[512]; f2.u = lp[1024]; f3.u = lp[1536];
        v16bf A0 = pack_bf16(u0a, u0b, u0c, u0d);
        v16bf A1 = pack_bf16(v0a, v0b, v0c, v0d);
        a00 = __builtin_amdgcn_wmma_f32_16x16x32_bf16(false, A0, false, f0.b, (short)0, a00, false, false);
        a01 = __builtin_amdgcn_wmma_f32_16x16x32_bf16(false, A0, false, f1.b, (short)0, a01, false, false);
        a02 = __builtin_amdgcn_wmma_f32_16x16x32_bf16(false, A0, false, f2.b, (short)0, a02, false, false);
        a03 = __builtin_amdgcn_wmma_f32_16x16x32_bf16(false, A0, false, f3.b, (short)0, a03, false, false);
        a10 = __builtin_amdgcn_wmma_f32_16x16x32_bf16(false, A1, false, f0.b, (short)0, a10, false, false);
        a11 = __builtin_amdgcn_wmma_f32_16x16x32_bf16(false, A1, false, f1.b, (short)0, a11, false, false);
        a12 = __builtin_amdgcn_wmma_f32_16x16x32_bf16(false, A1, false, f2.b, (short)0, a12, false, false);
        a13 = __builtin_amdgcn_wmma_f32_16x16x32_bf16(false, A1, false, f3.b, (short)0, a13, false, false);
      }

      // Prefetch step k2 into buffer 0.
      const float4* r0 = (const float4*)(cx0 + k2 * 32);
      const float4* r1 = (const float4*)(cx1 + k2 * 32);
      u0a = r0[0]; u0b = r0[1]; u0c = r0[4]; u0d = r0[5];
      v0a = r1[0]; v0b = r1[1]; v0c = r1[4]; v0d = r1[5];

      {  // Compute step k1 from buffer 1.
        const v16u* lp = ldsv + k1 * 32 + lane;
        BF16Frag f0, f1, f2, f3;
        f0.u = lp[0]; f1.u = lp[512]; f2.u = lp[1024]; f3.u = lp[1536];
        v16bf A0 = pack_bf16(u1a, u1b, u1c, u1d);
        v16bf A1 = pack_bf16(v1a, v1b, v1c, v1d);
        a00 = __builtin_amdgcn_wmma_f32_16x16x32_bf16(false, A0, false, f0.b, (short)0, a00, false, false);
        a01 = __builtin_amdgcn_wmma_f32_16x16x32_bf16(false, A0, false, f1.b, (short)0, a01, false, false);
        a02 = __builtin_amdgcn_wmma_f32_16x16x32_bf16(false, A0, false, f2.b, (short)0, a02, false, false);
        a03 = __builtin_amdgcn_wmma_f32_16x16x32_bf16(false, A0, false, f3.b, (short)0, a03, false, false);
        a10 = __builtin_amdgcn_wmma_f32_16x16x32_bf16(false, A1, false, f0.b, (short)0, a10, false, false);
        a11 = __builtin_amdgcn_wmma_f32_16x16x32_bf16(false, A1, false, f1.b, (short)0, a11, false, false);
        a12 = __builtin_amdgcn_wmma_f32_16x16x32_bf16(false, A1, false, f2.b, (short)0, a12, false, false);
        a13 = __builtin_amdgcn_wmma_f32_16x16x32_bf16(false, A1, false, f3.b, (short)0, a13, false, false);
      }
    }
  }

  // C/D layout: VGPR r, lanes 0-15 -> M=r, lanes 16-31 -> M=r+8; N = lane%16.
  float* hp0 = hpart + (size_t)kq * T_TOK * R_RANK
                     + (size_t)(mt0 * 16 + khalf * 8) * R_RANK + row;
  float* hp1 = hpart + (size_t)kq * T_TOK * R_RANK
                     + (size_t)(mt1 * 16 + khalf * 8) * R_RANK + row;
#pragma unroll
  for (int r = 0; r < 8; ++r) {
    hp0[r * R_RANK +  0] = a00[r];
    hp0[r * R_RANK + 16] = a01[r];
    hp0[r * R_RANK + 32] = a02[r];
    hp0[r * R_RANK + 48] = a03[r];
    hp1[r * R_RANK +  0] = a10[r];
    hp1[r * R_RANK + 16] = a11[r];
    hp1[r * R_RANK + 32] = a12[r];
    hp1[r * R_RANK + 48] = a13[r];
  }
}

// ---------------------------------------------------------------------------
// GEMM2: out[T,4096] = (h0+h1+h2+h3) @ b^T.  One wave per (16-token, 64-col)
// tile; sums the four K-quarter partials of h, then K = 64 -> 8 WMMAs.
// ---------------------------------------------------------------------------
__global__ __launch_bounds__(256) void gemm2_kernel(const float* __restrict__ hpart,
                                                    const unsigned short* __restrict__ bfrag,
                                                    float* __restrict__ out) {
  int task  = (blockIdx.x * 256 + threadIdx.x) >> 5;  // 0..65535
  int lane  = threadIdx.x & 31;
  int mt    = task >> 6;
  int nc    = task & 63;
  int row   = lane & 15;
  int khalf = lane >> 4;
  const size_t HS = (size_t)T_TOK * R_RANK;
  const float* hb = hpart + (size_t)(mt * 16 + row) * R_RANK + khalf * 8;
  const float4* q0 = (const float4*)hb;
  const float4* q1 = (const float4*)(hb + HS);
  const float4* q2 = (const float4*)(hb + 2 * HS);
  const float4* q3 = (const float4*)(hb + 3 * HS);
  const v16u* bfp = (const v16u*)bfrag + lane;

  float4 s0 = add4(add4(q0[0],  q1[0]),  add4(q2[0],  q3[0]));
  float4 s1 = add4(add4(q0[1],  q1[1]),  add4(q2[1],  q3[1]));
  float4 s2 = add4(add4(q0[4],  q1[4]),  add4(q2[4],  q3[4]));
  float4 s3 = add4(add4(q0[5],  q1[5]),  add4(q2[5],  q3[5]));
  float4 s4 = add4(add4(q0[8],  q1[8]),  add4(q2[8],  q3[8]));
  float4 s5 = add4(add4(q0[9],  q1[9]),  add4(q2[9],  q3[9]));
  float4 s6 = add4(add4(q0[12], q1[12]), add4(q2[12], q3[12]));
  float4 s7 = add4(add4(q0[13], q1[13]), add4(q2[13], q3[13]));

  v16u u00 = bfp[((nc * 4 + 0) * 2 + 0) * 32];
  v16u u01 = bfp[((nc * 4 + 1) * 2 + 0) * 32];
  v16u u02 = bfp[((nc * 4 + 2) * 2 + 0) * 32];
  v16u u03 = bfp[((nc * 4 + 3) * 2 + 0) * 32];
  v16u u10 = bfp[((nc * 4 + 0) * 2 + 1) * 32];
  v16u u11 = bfp[((nc * 4 + 1) * 2 + 1) * 32];
  v16u u12 = bfp[((nc * 4 + 2) * 2 + 1) * 32];
  v16u u13 = bfp[((nc * 4 + 3) * 2 + 1) * 32];

  v8f acc0 = {}, acc1 = {}, acc2 = {}, acc3 = {};

  v16bf A0 = pack_bf16(s0, s1, s2, s3);
  BF16Frag f0, f1, f2, f3;
  f0.u = u00; f1.u = u01; f2.u = u02; f3.u = u03;
  acc0 = __builtin_amdgcn_wmma_f32_16x16x32_bf16(false, A0, false, f0.b, (short)0, acc0, false, false);
  acc1 = __builtin_amdgcn_wmma_f32_16x16x32_bf16(false, A0, false, f1.b, (short)0, acc1, false, false);
  acc2 = __builtin_amdgcn_wmma_f32_16x16x32_bf16(false, A0, false, f2.b, (short)0, acc2, false, false);
  acc3 = __builtin_amdgcn_wmma_f32_16x16x32_bf16(false, A0, false, f3.b, (short)0, acc3, false, false);

  v16bf A1 = pack_bf16(s4, s5, s6, s7);
  f0.u = u10; f1.u = u11; f2.u = u12; f3.u = u13;
  acc0 = __builtin_amdgcn_wmma_f32_16x16x32_bf16(false, A1, false, f0.b, (short)0, acc0, false, false);
  acc1 = __builtin_amdgcn_wmma_f32_16x16x32_bf16(false, A1, false, f1.b, (short)0, acc1, false, false);
  acc2 = __builtin_amdgcn_wmma_f32_16x16x32_bf16(false, A1, false, f2.b, (short)0, acc2, false, false);
  acc3 = __builtin_amdgcn_wmma_f32_16x16x32_bf16(false, A1, false, f3.b, (short)0, acc3, false, false);

  float* op = out + (size_t)(mt * 16 + khalf * 8) * D_DIM + nc * 64 + row;
#pragma unroll
  for (int r = 0; r < 8; ++r) {
    op[(size_t)r * D_DIM +  0] = acc0[r];
    op[(size_t)r * D_DIM + 16] = acc1[r];
    op[(size_t)r * D_DIM + 32] = acc2[r];
    op[(size_t)r * D_DIM + 48] = acc3[r];
  }
}

extern "C" void kernel_launch(void* const* d_in, const int* in_sizes, int n_in,
                              void* d_out, int out_size, void* d_ws, size_t ws_size,
                              hipStream_t stream) {
  (void)in_sizes; (void)n_in; (void)out_size; (void)ws_size;
  const float* x         = (const float*)d_in[0];
  const int*   qa_vals   = (const int*)  d_in[1];
  const float* qa_scales = (const float*)d_in[2];
  const int*   qb_vals   = (const int*)  d_in[3];
  const float* qb_scales = (const float*)d_in[4];
  float* out = (float*)d_out;

  char* ws = (char*)d_ws;
  float*          hpart = (float*)ws;                          // 4 x 4 MB partial h
  unsigned short* afrag = (unsigned short*)(ws + 16777216);    // 512 KB
  unsigned short* bfrag = (unsigned short*)(ws + 17301504);    // 512 KB

  // Stream order serializes the pipeline: dequant -> gemm1 -> gemm2.
  deq_a_kernel<<<64, 256, 0, stream>>>(qa_vals, qa_scales, afrag);
  deq_b_kernel<<<64, 256, 0, stream>>>(qb_vals, qb_scales, bfrag);
  gemm1_kernel<<<512, 128, 0, stream>>>(x, afrag, hpart);
  gemm2_kernel<<<8192, 256, 0, stream>>>(hpart, bfrag, out);
}
